// BDGCN_11493332484865
// MI455X (gfx1250) — matrix operand order repaired and checked by reading, not compiled, not run on hardware
//
#include <hip/hip_runtime.h>

typedef __attribute__((ext_vector_type(16))) _Float16 v16h;
typedef __attribute__((ext_vector_type(8)))  _Float16 v8h;
typedef __attribute__((ext_vector_type(8)))  float    v8f;
typedef __attribute__((ext_vector_type(4)))  float    v4f_;
typedef __attribute__((ext_vector_type(4)))  unsigned v4u;
typedef __attribute__((ext_vector_type(8)))  int      v8i;
typedef __attribute__((ext_vector_type(4)))  int      v4i;

#define B_  8
#define N_  256
#define L_  32
#define K_  3
#define H_  64
#define CL_ 8192     /* N_*L_ */
#define KL_ 96       /* K_*L_ */

// workspace layout (bytes), all 256B aligned
#define GT_OFF 0u           /* GhT: 3*256*256 f16   = 393,216 B    */
#define XT_OFF 393216u      /* XhT: 8*8192*256 f16  = 33,554,432 B */
#define WT_OFF 33947648u    /* WhT: 3*64*96 f16     = 36,864 B     */
#define M1_OFF 33984512u    /* M1:  3*8*256*8192 f16= 100,663,296 B*/

#define HAVE_TDM (__has_builtin(__builtin_amdgcn_tensor_load_to_lds) && \
                  __has_builtin(__builtin_amdgcn_s_wait_tensorcnt))

static __device__ __forceinline__ v16h cat8(v8h lo, v8h hi) {
  return __builtin_shufflevector(lo, hi, 0, 1, 2, 3, 4, 5, 6, 7,
                                 8, 9, 10, 11, 12, 13, 14, 15);
}

// ---------------------------------------------------------------------------
// Setup A: tiled f32 -> f16 transpose.  dst[c][r] = (f16)src[r][c]
// ---------------------------------------------------------------------------
__global__ __launch_bounds__(256) void tpose_kernel(const float* __restrict__ src,
                                                    _Float16* __restrict__ dst,
                                                    int R, int C) {
  __shared__ _Float16 sT[32 * 72];
  const int tid = threadIdx.x;
  const size_t bo = (size_t)blockIdx.z * R * C;
  const int r0 = blockIdx.y * 32;
  const int c0 = blockIdx.x * 64;

  {  // read 32x64 f32 tile coalesced, convert, park in LDS
    const int rr = tid >> 3, cq = (tid & 7) * 8;
    const float* sp = src + bo + (size_t)(r0 + rr) * C + c0 + cq;
    v4f_ x0 = *(const v4f_*)(sp);
    v4f_ x1 = *(const v4f_*)(sp + 4);
    v8h o;
#pragma unroll
    for (int e = 0; e < 4; ++e) { o[e] = (_Float16)x0[e]; o[4 + e] = (_Float16)x1[e]; }
    *(v8h*)(&sT[rr * 72 + cq]) = o;
  }
  __syncthreads();
  {  // gather transposed rows, packed b128 store
    const int cc = tid >> 2, rq = (tid & 3) * 8;
    v8h o;
#pragma unroll
    for (int e = 0; e < 8; ++e) o[e] = sT[(rq + e) * 72 + cc];
    *(v8h*)(dst + bo + (size_t)(c0 + cc) * R + r0 + rq) = o;
  }
}

// ---------------------------------------------------------------------------
// Setup B: W[(k*K+j)*L+l][h] -> WhT[j][h][k*32+l]  (f16)
// ---------------------------------------------------------------------------
__global__ void wcvt_kernel(const float* __restrict__ W, _Float16* __restrict__ WhT) {
  unsigned idx = blockIdx.x * blockDim.x + threadIdx.x;
  if (idx < (unsigned)K_ * K_ * L_ * H_) {
    unsigned h   = idx % H_;
    unsigned row = idx / H_;
    unsigned l   = row % L_;
    unsigned kj  = row / L_;
    unsigned j   = kj % K_;
    unsigned k   = kj / K_;
    WhT[(j * H_ + h) * KL_ + k * L_ + l] = (_Float16)W[idx];
  }
}

// ---------------------------------------------------------------------------
// Kernel 1: D[cl][m] = sum_n XhT[b][cl][n] * GhT[k][m][n]  ( == m1[k,b,m,cl] )
// Block: 256 thr (8 waves) -> 128(cl, M-dim) x 64(m, N-dim) tile.
// Whole 64x256 B panel (32KB) pulled once by the Tensor Data Mover into LDS
// with hardware row padding (528B rows); n0 loop runs barrier-free.
// ---------------------------------------------------------------------------
__global__ __launch_bounds__(256) void stage1_kernel(const _Float16* __restrict__ GhT,
                                                     const _Float16* __restrict__ XhT,
                                                     _Float16* __restrict__ M1) {
  __shared__ _Float16 sB[64 * 264];  // GhT panel [m][n], 528B rows (16B aligned)
  const int tid  = threadIdx.x;
  const int lane = tid & 31;
  const int w    = tid >> 5;
  const int cl0  = blockIdx.x * 128;
  const int m0   = blockIdx.y * 64;
  const int k    = blockIdx.z >> 3;
  const int b    = blockIdx.z & 7;

  const int rlo = lane & 15;
  const int kbA = (lane & 16) >> 1;  // 0 or 8
  const int kbB = lane & 16;         // 0 or 16
  const int mhi = (lane >> 4) << 3;

  const _Float16* Gk = GhT + (size_t)k * N_ * N_ + (size_t)m0 * N_;

#if HAVE_TDM
  if (w == 0) {
    // Tensor DMA descriptor (D#): 2D tile 256(n) x 64(m rows), 2B elements,
    // LDS pad 16B after every 512B row -> 264-half row stride in sB.
    const unsigned lds = (unsigned)(uintptr_t)(&sB[0]);
    const unsigned long long ga = (unsigned long long)(uintptr_t)Gk;
    v4u g0 = {1u,                                   // count=1, user mode
              lds,                                  // lds_addr
              (unsigned)ga,                         // global_addr[31:0]
              (unsigned)((ga >> 32) & 0x1FFFFFFu) | (2u << 30)};  // addr[56:32], type=2
    v8i g1 = {(int)((1u << 16)                      // data_size = 2B
                    | (1u << 20)                    // pad_enable
                    | (6u << 22)                    // pad_interval: 128 DWORDs (512B)
                    | (3u << 25)),                  // pad_amount: 4 DWORDs (16B)
              (int)(256u << 16),                    // tensor_dim0 = 256  [31:16]=lo16
              (int)(64u << 16),                     // dim0 hi=0 | tensor_dim1 lo16 = 64
              (int)(256u << 16),                    // dim1 hi=0 | tile_dim0 = 256
              (int)64u,                             // tile_dim1 = 64 | tile_dim2 = 0
              (int)256u,                            // tensor_dim0_stride lo32 = 256
              0, 0};                                // stride hi | tensor_dim1_stride = 0
    v4i gz4 = {0, 0, 0, 0};
    v8i gz8 = {0, 0, 0, 0, 0, 0, 0, 0};
    __builtin_amdgcn_tensor_load_to_lds(g0, g1, gz4, gz4, gz8, 0);
    __builtin_amdgcn_s_wait_tensorcnt(0);
  }
#else
  {  // fallback: vectorized cooperative copy of the 64x256 panel
    const int mm = tid >> 2, nq0 = (tid & 3) * 64;
#pragma unroll
    for (int q = 0; q < 8; ++q) {
      const int nq = nq0 + q * 8;
      *(v8h*)(&sB[mm * 264 + nq]) = *(const v8h*)(&Gk[(size_t)mm * N_ + nq]);
    }
  }
#endif
  __syncthreads();

  const v8f zz = {0.f, 0.f, 0.f, 0.f, 0.f, 0.f, 0.f, 0.f};
  v8f acc[4];
#pragma unroll
  for (int s = 0; s < 4; ++s) acc[s] = zz;

  const _Float16* Arow0 = XhT + ((size_t)b * CL_ + cl0 + w * 16 + rlo) * N_;

  for (int n0 = 0; n0 < N_; n0 += 32) {
    const _Float16* arow = Arow0 + n0;
    v16h a = cat8(*(const v8h*)(arow + kbA), *(const v8h*)(arow + 16 + kbA));
#pragma unroll
    for (int s = 0; s < 4; ++s) {
      const _Float16* brow = &sB[(s * 16 + rlo) * 264 + n0 + kbA];
      v16h bf = cat8(*(const v8h*)(brow), *(const v8h*)(brow + 16));
      acc[s] = __builtin_amdgcn_wmma_f32_16x16x32_f16(false, a, false, bf,
                                                      (short)0, acc[s], false, false);
    }
  }

  // packed store: 8 consecutive cl per lane
  _Float16* M1p = M1 + ((size_t)k * B_ + b) * N_ * CL_;
#pragma unroll
  for (int s = 0; s < 4; ++s) {
    const int m = m0 + s * 16 + rlo;
    v8h pv;
#pragma unroll
    for (int r = 0; r < 8; ++r) pv[r] = (_Float16)acc[s][r];
    *(v8h*)(&M1p[(size_t)m * CL_ + cl0 + w * 16 + mhi]) = pv;
  }
}

// ---------------------------------------------------------------------------
// Kernel 2: per (b,m):
//   GEMM1: P[c][h]   = sum_{k,l} m1[k,b,m,c,l] * W[(kK+j)L+l, h]  -> sPT[h][c]
//   GEMM2: Ht[h][d] += sum_c sPT[h][c] * GhT[j][d][c]
//   out[b][m][d][h] = relu(Ht + bias)  (packed f32 stores)
// ---------------------------------------------------------------------------
__global__ __launch_bounds__(256) void stage23_kernel(const _Float16* __restrict__ M1,
                                                      const _Float16* __restrict__ GhT,
                                                      const _Float16* __restrict__ WhT,
                                                      const float* __restrict__ bias,
                                                      float* __restrict__ out) {
  __shared__ _Float16 sPT[64 * 264];  // P transposed [h][c], 528B rows
  const int tid  = threadIdx.x;
  const int lane = tid & 31;
  const int w    = tid >> 5;
  const int m    = blockIdx.x;
  const int b    = blockIdx.y;

  __builtin_prefetch(WhT, 0, 3);  // global_prefetch_b8: W is hot for every block

  const int rlo = lane & 15;
  const int kbA = (lane & 16) >> 1;
  const int kbB = lane & 16;
  const int mhi = (lane >> 4) << 3;

  // GEMM2 wave mapping: h-tile = w&3 (M dim, 4 tiles), d-tiles (w>>2)*8 + 0..7
  const int ht2   = w & 3;
  const int dbase = (w >> 2) * 8;

  const v8f zz = {0.f, 0.f, 0.f, 0.f, 0.f, 0.f, 0.f, 0.f};
  v8f hacc[8];
#pragma unroll
  for (int dt = 0; dt < 8; ++dt) hacc[dt] = zz;

  for (int j = 0; j < K_; ++j) {
    __syncthreads();  // sPT reuse guard across j iterations

    // ---- GEMM1: P = m1_slice[256 x 96] @ W_j[96 x 64] ----
    v8f pacc[2][4];
#pragma unroll
    for (int ct = 0; ct < 2; ++ct)
#pragma unroll
      for (int ht = 0; ht < 4; ++ht) pacc[ct][ht] = zz;

#pragma unroll
    for (int ks = 0; ks < 3; ++ks) {  // K-step == support index k (kdim = k*32+l)
      const _Float16* m1p = M1 + (((size_t)ks * B_ + b) * N_ + m) * CL_;
      v16h a[2];
#pragma unroll
      for (int ct = 0; ct < 2; ++ct) {
        const _Float16* arow = m1p + (size_t)((w + ct * 8) * 16 + rlo) * L_;
        a[ct] = cat8(*(const v8h*)(arow + kbA), *(const v8h*)(arow + 16 + kbA));
      }
#pragma unroll
      for (int ht = 0; ht < 4; ++ht) {
        const _Float16* brow =
            WhT + (size_t)(j * H_ + ht * 16 + rlo) * KL_ + ks * 32 + kbA;
        v16h bf = cat8(*(const v8h*)(brow), *(const v8h*)(brow + 16));
#pragma unroll
        for (int ct = 0; ct < 2; ++ct)
          pacc[ct][ht] = __builtin_amdgcn_wmma_f32_16x16x32_f16(
              false, a[ct], false, bf, (short)0, pacc[ct][ht], false, false);
      }
    }
    // spill P transposed: per lane 8 consecutive c -> one packed 16B store
#pragma unroll
    for (int ct = 0; ct < 2; ++ct)
#pragma unroll
      for (int ht = 0; ht < 4; ++ht) {
        const int h  = ht * 16 + rlo;
        const int c0 = (w + ct * 8) * 16 + mhi;
        v8h pv;
#pragma unroll
        for (int r = 0; r < 8; ++r) pv[r] = (_Float16)pacc[ct][ht][r];
        *(v8h*)(&sPT[h * 264 + c0]) = pv;
      }
    __syncthreads();

    // ---- GEMM2 (flipped): Ht[h][d] += P^T[h][c] * G[j][c][d] ----
    const _Float16* Gj = GhT + (size_t)j * N_ * N_;
    for (int cs = 0; cs < 8; ++cs) {
      // A fragment from sPT row (one per wave per cs, reused over 8 d-tiles)
      const _Float16* prow = &sPT[(ht2 * 16 + rlo) * 264 + cs * 32];
      v16h a2 = cat8(*(const v8h*)(prow + kbA), *(const v8h*)(prow + 16 + kbA));
#pragma unroll
      for (int dt = 0; dt < 8; ++dt) {
        const int d = (dbase + dt) * 16 + rlo;
        const _Float16* brow = Gj + (size_t)d * N_ + cs * 32 + kbA;
        v16h bf = cat8(*(const v8h*)(brow), *(const v8h*)(brow + 16));
        hacc[dt] = __builtin_amdgcn_wmma_f32_16x16x32_f16(
            false, a2, false, bf, (short)0, hacc[dt], false, false);
      }
    }
  }

  // ---- epilogue: bias + relu, packed f32 stores (8 consecutive h per lane) ----
  const int h0 = ht2 * 16 + mhi;
  const v4f_ bb0 = *(const v4f_*)(bias + h0);
  const v4f_ bb1 = *(const v4f_*)(bias + h0 + 4);
  float* op = out + ((size_t)b * N_ + m) * N_ * H_;
#pragma unroll
  for (int dt = 0; dt < 8; ++dt) {
    const int d = (dbase + dt) * 16 + rlo;
    v4f_ o0, o1;
#pragma unroll
    for (int r = 0; r < 4; ++r) {
      float v0 = hacc[dt][r] + bb0[r];
      float v1 = hacc[dt][4 + r] + bb1[r];
      o0[r] = v0 > 0.f ? v0 : 0.f;
      o1[r] = v1 > 0.f ? v1 : 0.f;
    }
    float* dp = op + (size_t)d * H_ + h0;
    *(v4f_*)(dp)     = o0;
    *(v4f_*)(dp + 4) = o1;
  }
}

// ---------------------------------------------------------------------------
extern "C" void kernel_launch(void* const* d_in, const int* in_sizes, int n_in,
                              void* d_out, int out_size, void* d_ws, size_t ws_size,
                              hipStream_t stream) {
  const float* X  = (const float*)d_in[0];
  const float* G  = (const float*)d_in[1];
  const float* W  = (const float*)d_in[2];
  const float* bs = (const float*)d_in[3];
  float* out = (float*)d_out;

  char* ws = (char*)d_ws;
  _Float16* GhT = (_Float16*)(ws + GT_OFF);
  _Float16* XhT = (_Float16*)(ws + XT_OFF);
  _Float16* WhT = (_Float16*)(ws + WT_OFF);
  _Float16* M1  = (_Float16*)(ws + M1_OFF);

  (void)in_sizes; (void)n_in; (void)out_size; (void)ws_size;

  // X[b][n][cl] -> XhT[b][cl][n] ; G[k][n][m] -> GhT[k][m][n]
  tpose_kernel<<<dim3(CL_ / 64, N_ / 32, B_), 256, 0, stream>>>(X, XhT, N_, CL_);
  tpose_kernel<<<dim3(N_ / 64, N_ / 32, K_), 256, 0, stream>>>(G, GhT, N_, N_);
  wcvt_kernel<<<(K_ * K_ * L_ * H_ + 255) / 256, 256, 0, stream>>>(W, WhT);

  stage1_kernel<<<dim3(CL_ / 128, N_ / 64, K_ * B_), 256, 0, stream>>>(GhT, XhT, M1);
  stage23_kernel<<<dim3(N_, B_), 256, 0, stream>>>(M1, GhT, WhT, bs, out);
}
